// EdgeDecoder_71914932404373
// MI455X (gfx1250) — compile-verified, matrix operand order checked by reading, not run on hardware
//
#include <hip/hip_runtime.h>
#include <hip/hip_bf16.h>

// ---- CDNA5 vector types ----
typedef __attribute__((ext_vector_type(16))) __bf16 v16bf;
typedef __attribute__((ext_vector_type(8)))  __bf16 v8bf;
typedef __attribute__((ext_vector_type(8)))  float  v8f;
typedef __attribute__((ext_vector_type(4)))  float  v4f;
typedef __attribute__((ext_vector_type(8)))  float  v8fv;

#define Z_DIM    64
#define EATTR    32
#define IN_DIM   160          // 2*Z_DIM + EATTR
#define HID      128
#define KSTEPS   5            // 160 / 32
#define NTILES   8            // 128 / 16
#define MT       2            // 2 x 16 edges per wave-tile
#define WSTRIDE  168          // padded LDS row stride (halves) -> conflict-free

// load 8 consecutive fp32, convert to 8 bf16
__device__ __forceinline__ v8bf cvt8(const float* __restrict__ p) {
    v4f a = *(const v4f*)p;
    v4f b = *(const v4f*)(p + 4);
    v8fv f = __builtin_shufflevector(a, b, 0, 1, 2, 3, 4, 5, 6, 7);
    return __builtin_convertvector(f, v8bf);
}

// pointer to element k of the virtual 160-dim input row of edge e.
// k is a compile-time-known multiple of 8 per unrolled k-step, so the
// branch below constant-folds (each 8-chunk lies wholly in one source).
__device__ __forceinline__ const float* xptr(const float* __restrict__ z,
                                             const float* __restrict__ ea,
                                             int src, int dst, int e, int k) {
    if (k < Z_DIM)       return z  + (size_t)src * Z_DIM + k;
    if (k < 2 * Z_DIM)   return z  + (size_t)dst * Z_DIM + (k - Z_DIM);
    return ea + (size_t)e * EATTR + (k - 2 * Z_DIM);
}

__global__ __launch_bounds__(256) void edge_mlp_wmma(
    const float* __restrict__ z,   // [N_NODES, 64]
    const int*   __restrict__ ei,  // [2, E]
    const float* __restrict__ ea,  // [E, 32]
    const float* __restrict__ W1,  // [160, 128]
    const float* __restrict__ b1,  // [128]
    const float* __restrict__ W2,  // [128]
    const float* __restrict__ b2,  // [1]
    float*       __restrict__ out, // [E]
    int nE)
{
    // W1 transposed to bf16 in LDS: Wt[n][k], padded stride avoids bank conflicts
    __shared__ __align__(16) __bf16 Wt[HID * WSTRIDE];

    const int tid = threadIdx.x;
    for (int idx = tid; idx < IN_DIM * HID; idx += 256) {
        int k = idx >> 7;          // 0..159
        int n = idx & (HID - 1);   // 0..127
        Wt[n * WSTRIDE + k] = (__bf16)W1[idx];
    }
    __syncthreads();

    const int lane  = tid & 31;
    const int wid   = tid >> 5;
    const int lhalf = lane >> 4;   // 0: lanes 0-15, 1: lanes 16-31
    const int l16   = lane & 15;

    // per-lane epilogue constants: n = nt*16 + l16
    float b1l[NTILES], w2l[NTILES];
#pragma unroll
    for (int nt = 0; nt < NTILES; ++nt) {
        b1l[nt] = b1[nt * 16 + l16];
        w2l[nt] = W2[nt * 16 + l16];
    }
    const float b2v = b2[0];

    const long long gw     = (long long)blockIdx.x * 8 + wid;
    const long long nw     = (long long)gridDim.x * 8;
    const long long ntiles = (nE + (16 * MT - 1)) / (16 * MT);

    for (long long t = gw; t < ntiles; t += nw) {
        const int e0 = (int)(t * 16 * MT);

        // --- gather edge endpoints for this lane's rows (clamped, EXEC stays full) ---
        int eSrc[MT], eDst[MT], eRow[MT];
#pragma unroll
        for (int m = 0; m < MT; ++m) {
            int e = e0 + m * 16 + l16;
            if (e >= nE) e = nE - 1;
            eRow[m] = e;
            eSrc[m] = ei[e];
            eDst[m] = ei[(size_t)nE + e];
        }

        // --- build bf16 A fragments: afrag[m][ks] = rows of [z_src|z_dst|eattr] ---
        // 16-bit A layout: lanes 0-15 hold K = {k0..k0+7, k0+16..k0+23} with k0 = ks*32,
        // lanes 16-31 the same +8.
        v16bf afrag[MT][KSTEPS];
#pragma unroll
        for (int ks = 0; ks < KSTEPS; ++ks) {
            const int k0 = ks * 32 + lhalf * 8;
#pragma unroll
            for (int m = 0; m < MT; ++m) {
                v8bf lo = cvt8(xptr(z, ea, eSrc[m], eDst[m], eRow[m], k0));
                v8bf hi = cvt8(xptr(z, ea, eSrc[m], eDst[m], eRow[m], k0 + 16));
                afrag[m][ks] = __builtin_shufflevector(
                    lo, hi, 0, 1, 2, 3, 4, 5, 6, 7, 8, 9, 10, 11, 12, 13, 14, 15);
            }
        }

        float part[MT][8];
#pragma unroll
        for (int m = 0; m < MT; ++m)
#pragma unroll
            for (int r = 0; r < 8; ++r) part[m][r] = 0.f;

        // --- layer 1 GEMM tiles + fused layer-2 epilogue ---
#pragma unroll
        for (int nt = 0; nt < NTILES; ++nt) {
            v8f acc[MT];
#pragma unroll
            for (int m = 0; m < MT; ++m) acc[m] = (v8f){};

#pragma unroll
            for (int ks = 0; ks < KSTEPS; ++ks) {
                // B fragment from LDS: row n = nt*16 + l16, mirrored K layout
                const __bf16* bp = &Wt[(nt * 16 + l16) * WSTRIDE + ks * 32 + lhalf * 8];
                v8bf blo = *(const v8bf*)bp;
                v8bf bhi = *(const v8bf*)(bp + 16);
                v16bf bfrag = __builtin_shufflevector(
                    blo, bhi, 0, 1, 2, 3, 4, 5, 6, 7, 8, 9, 10, 11, 12, 13, 14, 15);
#pragma unroll
                for (int m = 0; m < MT; ++m) {
                    acc[m] = __builtin_amdgcn_wmma_f32_16x16x32_bf16(
                        false, afrag[m][ks], false, bfrag,
                        (short)0, acc[m], false, false);
                }
            }

            // epilogue: h = relu(acc + b1[n]); part += h * W2[n]
#pragma unroll
            for (int m = 0; m < MT; ++m) {
#pragma unroll
                for (int r = 0; r < 8; ++r) {
                    float h = acc[m][r] + b1l[nt];
                    h = h > 0.f ? h : 0.f;
                    part[m][r] += h * w2l[nt];
                }
            }
        }

        // --- reduce across the 16 lanes of each half (N dimension) ---
        // C layout: VGPR r, lanes 0-15 -> row M=r; lanes 16-31 -> row M=r+8
#pragma unroll
        for (int m = 0; m < MT; ++m) {
#pragma unroll
            for (int r = 0; r < 8; ++r) {
                float v = part[m][r];
                v += __shfl_xor(v, 1, 32);
                v += __shfl_xor(v, 2, 32);
                v += __shfl_xor(v, 4, 32);
                v += __shfl_xor(v, 8, 32);
                part[m][r] = v;
            }
        }

        if (l16 == 0) {
#pragma unroll
            for (int m = 0; m < MT; ++m) {
                const int eb = e0 + m * 16 + lhalf * 8;
#pragma unroll
                for (int r = 0; r < 8; ++r) {
                    if (eb + r < nE) out[eb + r] = part[m][r] + b2v;
                }
            }
        }
    }
}

extern "C" void kernel_launch(void* const* d_in, const int* in_sizes, int n_in,
                              void* d_out, int out_size, void* d_ws, size_t ws_size,
                              hipStream_t stream) {
    const float* z  = (const float*)d_in[0];
    const int*   ei = (const int*)d_in[1];
    const float* ea = (const float*)d_in[2];
    const float* W1 = (const float*)d_in[3];
    const float* b1 = (const float*)d_in[4];
    const float* W2 = (const float*)d_in[5];
    const float* b2 = (const float*)d_in[6];
    float* out = (float*)d_out;
    const int nE = out_size;

    dim3 grid(1024), block(256);
    hipLaunchKernelGGL(edge_mlp_wmma, grid, block, 0, stream,
                       z, ei, ea, W1, b1, W2, b2, out, nE);
}
